// Attention_31104153158132
// MI455X (gfx1250) — matrix operand order repaired and checked by reading, not compiled, not run on hardware
//
#include <hip/hip_runtime.h>
#include <hip/hip_bf16.h>
#include <math.h>

#define BS_ 8
#define S_ 128
#define H_ 512
#define NH_ 8
#define DH_ 64
#define SS_ (S_ * S_)
#define MTOT_ (BS_ * SS_)   /* 131072 edge rows */
#define MX_ (BS_ * S_)      /* 1024 token rows  */
#define LDA1 520            /* padded LDS row stride (bf16 elems) for H=512 */

typedef __bf16 bf16;
typedef __attribute__((ext_vector_type(16))) __bf16 v16bf;
typedef __attribute__((ext_vector_type(8)))  float  v8f;

union FragU { v16bf v; unsigned u[8]; };

// ---------------------------------------------------------------------------
// WMMA fragment helpers (layouts per CDNA5 ISA 7.12.2, wave32)
// ---------------------------------------------------------------------------

// A fragment: 16x32 bf16 tile, A row-major with stride lda.
// lanes 0-15: row m=lane, K pairs {0..7,16..23}; lanes 16-31: K pairs {8..15,24..31}
__device__ __forceinline__ v16bf load_frag_a(const bf16* A, int lda, int lane) {
  FragU f;
  int m  = lane & 15;
  int kb = (lane >> 4) << 3;            // 0 or 8
  const bf16* p = A + m * lda + kb;
#pragma unroll
  for (int j = 0; j < 4; ++j) {
    f.u[j]     = *(const unsigned*)(p + 2 * j);
    f.u[j + 4] = *(const unsigned*)(p + 16 + 2 * j);
  }
  return f.v;
}

// B fragment: 32(K)x16(N) bf16 tile, sourced from transposed matrix Bt
// (rows = output columns, contiguous K, stride ldb).
// lanes 0-15: col n=lane, K {0..7,16..23}; lanes 16-31: K {8..15,24..31}
__device__ __forceinline__ v16bf load_frag_b(const bf16* Bt, int ldb, int lane) {
  FragU f;
  int n  = lane & 15;
  int kb = (lane >> 4) << 3;            // 0 or 8
  const bf16* p = Bt + n * ldb + kb;
#pragma unroll
  for (int j = 0; j < 4; ++j) {
    f.u[j]     = *(const unsigned*)(p + 2 * j);
    f.u[j + 4] = *(const unsigned*)(p + 16 + 2 * j);
  }
  return f.v;
}

__device__ __forceinline__ v8f wmma_bf(v16bf a, v16bf b, v8f c) {
  return __builtin_amdgcn_wmma_f32_16x16x32_bf16(false, a, false, b, (short)0, c,
                                                 false, false);
}

// ---------------------------------------------------------------------------
// Small utility kernels
// ---------------------------------------------------------------------------

// Wt[n*K + k] = bf16(W[k*N + n])   (transpose + convert)
__global__ void wconv_kernel(const float* __restrict__ W, bf16* __restrict__ Wt,
                             int K, int N) {
  int idx = blockIdx.x * 256 + threadIdx.x;
  if (idx >= N * K) return;
  int n = idx / K, kk = idx - n * K;
  Wt[idx] = (bf16)W[kk * N + n];
}

__global__ void cast_kernel(const float* __restrict__ src, bf16* __restrict__ dst,
                            int n) {
  int idx = blockIdx.x * 256 + threadIdx.x;
  if (idx < n) dst[idx] = (bf16)src[idx];
}

// fp32 [BS*S, H] -> bf16 head layout. transposed=0 : [b,h,s,d] ; =1 : [b,h,d,s]
__global__ void qkv_reshape_kernel(const float* __restrict__ t, bf16* __restrict__ dst,
                                   float scale, int transposed) {
  int idx = blockIdx.x * 256 + threadIdx.x;
  if (idx >= MX_ * H_) return;
  int bs = idx / H_, col = idx - bs * H_;
  int b = bs / S_, s = bs - b * S_;
  int h = col / DH_, d = col - h * DH_;
  float v = t[idx] * scale;
  long base = (long)(b * NH_ + h);
  if (!transposed) dst[(base * S_ + s) * DH_ + d] = (bf16)v;
  else             dst[(base * DH_ + d) * S_ + s] = (bf16)v;
}

// ---------------------------------------------------------------------------
// Generic bf16 GEMM: C = act(A[M,K] @ Bt[N,K]^T + bias), fp32 accumulate.
// 256 threads = 8 waves (4x2), tile 128x128, wave tile 32x64. K % 32 == 0.
// act: 0 none, 1 relu, 2 exact gelu
// ---------------------------------------------------------------------------
__global__ void gemm_bf16_kernel(const bf16* __restrict__ A, const bf16* __restrict__ Bt,
                                 const float* __restrict__ bias,
                                 float* __restrict__ Cf, bf16* __restrict__ Cb,
                                 int M, int N, int K, int act) {
  int lane = threadIdx.x & 31;
  int wave = threadIdx.x >> 5;
  int bm = blockIdx.y * 128 + (wave >> 1) * 32;
  int bn = blockIdx.x * 128 + (wave & 1) * 64;

  v8f acc[2][4] = {};
  for (int k0 = 0; k0 < K; k0 += 32) {
    v16bf a0 = load_frag_a(A + (long)bm * K + k0, K, lane);
    v16bf a1 = load_frag_a(A + (long)(bm + 16) * K + k0, K, lane);
#pragma unroll
    for (int nt = 0; nt < 4; ++nt) {
      v16bf b = load_frag_b(Bt + (long)(bn + nt * 16) * K + k0, K, lane);
      acc[0][nt] = wmma_bf(a0, b, acc[0][nt]);
      acc[1][nt] = wmma_bf(a1, b, acc[1][nt]);
    }
  }

  int nl = lane & 15, mb = (lane >> 4) << 3;
#pragma unroll
  for (int mt = 0; mt < 2; ++mt)
#pragma unroll
    for (int nt = 0; nt < 4; ++nt)
#pragma unroll
      for (int j = 0; j < 8; ++j) {
        int m = bm + mt * 16 + mb + j;
        int n = bn + nt * 16 + nl;
        float v0 = acc[mt][nt][j];
        if (bias) v0 += bias[n];
        if (act == 1) v0 = fmaxf(v0, 0.0f);
        else if (act == 2) v0 = 0.5f * v0 * (1.0f + erff(v0 * 0.70710678118654752f));
        long o = (long)m * N + n;
        if (Cf) Cf[o] = v0;
        if (Cb) Cb[o] = (bf16)v0;
      }
}

// ---------------------------------------------------------------------------
// Fused edge MLP: coeffs = (relu(relu(E@W1+b1)@W2+b2))@W3+b3
// One workgroup = 128 rows. Hidden activations stay in LDS (2 x 128x520 bf16).
// edge_attr is read from HBM exactly once.
// ---------------------------------------------------------------------------
__device__ __forceinline__ void tile_gemm_relu_512(const bf16* Asrc, const bf16* Wt,
                                                   const float* __restrict__ bias,
                                                   bf16* Dst, int lane, int wave) {
  int cn = wave * 64;   // 8 waves x 64 cols = 512
#pragma unroll 1
  for (int mg = 0; mg < 2; ++mg) {
    int rm = mg * 64;
    v8f acc[4][4] = {};
    for (int k0 = 0; k0 < 512; k0 += 32) {
      v16bf bfr[4];
#pragma unroll
      for (int nt = 0; nt < 4; ++nt)
        bfr[nt] = load_frag_b(Wt + (cn + nt * 16) * 512 + k0, 512, lane);
#pragma unroll
      for (int mt = 0; mt < 4; ++mt) {
        v16bf a = load_frag_a(Asrc + (rm + mt * 16) * LDA1 + k0, LDA1, lane);
#pragma unroll
        for (int nt = 0; nt < 4; ++nt)
          acc[mt][nt] = wmma_bf(a, bfr[nt], acc[mt][nt]);
      }
    }
    int nl = lane & 15, mb = (lane >> 4) << 3;
#pragma unroll
    for (int mt = 0; mt < 4; ++mt)
#pragma unroll
      for (int nt = 0; nt < 4; ++nt)
#pragma unroll
        for (int j = 0; j < 8; ++j) {
          int m = rm + mt * 16 + mb + j;
          int n = cn + nt * 16 + nl;
          float v0 = fmaxf(acc[mt][nt][j] + bias[n], 0.0f);
          Dst[m * LDA1 + n] = (bf16)v0;
        }
  }
}

__global__ void edge_mlp_kernel(const float* __restrict__ Ein,
                                const bf16* __restrict__ W1t, const float* __restrict__ b1,
                                const bf16* __restrict__ W2t, const float* __restrict__ b2,
                                const float* __restrict__ W3, const float* __restrict__ b3,
                                float* __restrict__ coeffs) {
  extern __shared__ char smem[];
  bf16* bufA = (bf16*)smem;                               // 128 x LDA1
  bf16* h1   = (bf16*)(smem + 128 * LDA1 * sizeof(bf16)); // 128 x LDA1
  int tid = threadIdx.x;
  long row0 = (long)blockIdx.x * 128;

  // Phase A: stream-convert 128x512 fp32 tile -> bf16 LDS
  const float* src = Ein + row0 * 512;
  for (int i = tid * 4; i < 128 * 512; i += 256 * 4) {
    int r = i >> 9, c = i & 511;
    float4 f = *(const float4*)(src + i);
    bf16* d = bufA + r * LDA1 + c;
    d[0] = (bf16)f.x; d[1] = (bf16)f.y; d[2] = (bf16)f.z; d[3] = (bf16)f.w;
  }
  __syncthreads();

  int lane = tid & 31, wave = tid >> 5;
  tile_gemm_relu_512(bufA, W1t, b1, h1, lane, wave);   // layer 1
  __syncthreads();
  tile_gemm_relu_512(h1, W2t, b2, bufA, lane, wave);   // layer 2 (h2 overwrites input tile)
  __syncthreads();

  // layer 3: [128,512] @ [512,8] via VALU dot products (tiny)
  for (int i = tid; i < 128 * 8; i += 256) {
    int r = i >> 3, nh = i & 7;
    float s = b3[nh];
    for (int kk = 0; kk < 512; ++kk)
      s += (float)bufA[r * LDA1 + kk] * W3[kk * 8 + nh];
    coeffs[(row0 + r) * 8 + nh] = s;
  }
}

// ---------------------------------------------------------------------------
// Fused attention per (b,h): scores WMMA -> alpha*s + beta*bias -> softmax ->
// probs(bf16) @ vT WMMA. One workgroup per head (64 total), 256 threads.
// ---------------------------------------------------------------------------
#define SC_LD 132
#define PB_LD 136
__global__ void attention_kernel(const bf16* __restrict__ q, const bf16* __restrict__ k,
                                 const bf16* __restrict__ vT,
                                 const float* __restrict__ coeffs,
                                 const float* __restrict__ alpha_p,
                                 const float* __restrict__ beta_p,
                                 bf16* __restrict__ attn_out) {
  extern __shared__ char smem[];
  float* sc = (float*)smem;                              // 128 x SC_LD fp32
  bf16*  pb = (bf16*)(smem + 128 * SC_LD * sizeof(float)); // 128 x PB_LD bf16
  int tid = threadIdx.x, lane = tid & 31, wave = tid >> 5;
  int b = blockIdx.x >> 3, h = blockIdx.x & 7;
  const bf16* qh = q  + (long)(b * NH_ + h) * S_ * DH_;
  const bf16* kh = k  + (long)(b * NH_ + h) * S_ * DH_;
  const bf16* vh = vT + (long)(b * NH_ + h) * DH_ * S_;
  float alpha = *alpha_p, beta = *beta_p;

  // scores: wave w owns query rows [16w, 16w+16), all 128 key columns
  v8f acc[8] = {};
  for (int kc = 0; kc < DH_; kc += 32) {
    v16bf a = load_frag_a(qh + (16 * wave) * DH_ + kc, DH_, lane);
#pragma unroll
    for (int nt = 0; nt < 8; ++nt) {
      v16bf bb = load_frag_b(kh + (nt * 16) * DH_ + kc, DH_, lane);
      acc[nt] = wmma_bf(a, bb, acc[nt]);
    }
  }
  int nl = lane & 15, mb = (lane >> 4) << 3;
#pragma unroll
  for (int nt = 0; nt < 8; ++nt)
#pragma unroll
    for (int j = 0; j < 8; ++j) {
      int m = 16 * wave + mb + j;
      int n = nt * 16 + nl;
      float bias = coeffs[((long)b * SS_ + (long)m * S_ + n) * NH_ + h];
      sc[m * SC_LD + n] = alpha * acc[nt][j] + beta * bias;
    }
  __syncthreads();

  // softmax, one thread per query row
  if (tid < S_) {
    float mx = -3.4e38f;
    for (int c = 0; c < S_; ++c) mx = fmaxf(mx, sc[tid * SC_LD + c]);
    float sum = 0.0f;
    for (int c = 0; c < S_; ++c) {
      float e = expf(sc[tid * SC_LD + c] - mx);
      sc[tid * SC_LD + c] = e;
      sum += e;
    }
    float inv = 1.0f / sum;
    for (int c = 0; c < S_; ++c)
      pb[tid * PB_LD + c] = (bf16)(sc[tid * SC_LD + c] * inv);
  }
  __syncthreads();

  // out = probs @ v : M=128(q), N=64(d), K=128(keys)
  v8f oacc[4] = {};
  for (int kc = 0; kc < S_; kc += 32) {
    v16bf a = load_frag_a(pb + (16 * wave) * PB_LD + kc, PB_LD, lane);
#pragma unroll
    for (int nt = 0; nt < 4; ++nt) {
      v16bf bb = load_frag_b(vh + (nt * 16) * S_ + kc, S_, lane);
      oacc[nt] = wmma_bf(a, bb, oacc[nt]);
    }
  }
#pragma unroll
  for (int nt = 0; nt < 4; ++nt)
#pragma unroll
    for (int j = 0; j < 8; ++j) {
      int m = 16 * wave + mb + j;
      int d = nt * 16 + nl;
      attn_out[((long)b * S_ + m) * H_ + h * DH_ + d] = (bf16)oacc[nt][j];
    }
}

// ---------------------------------------------------------------------------
// LayerNorm: out = LN(a + res) * g + b. One block (128 thr) per row, H=512.
// ---------------------------------------------------------------------------
__global__ void ln_kernel(const float* __restrict__ a, const float* __restrict__ res,
                          const float* __restrict__ g, const float* __restrict__ bta,
                          float* __restrict__ outf, bf16* __restrict__ outb) {
  __shared__ float red[128];
  int r = blockIdx.x, t = threadIdx.x;
  float v[4], s = 0.0f;
#pragma unroll
  for (int i = 0; i < 4; ++i) {
    int c = t + 128 * i;
    v[i] = a[(long)r * H_ + c] + res[(long)r * H_ + c];
    s += v[i];
  }
  red[t] = s; __syncthreads();
  for (int off = 64; off > 0; off >>= 1) {
    if (t < off) red[t] += red[t + off];
    __syncthreads();
  }
  float mu = red[0] * (1.0f / H_);
  __syncthreads();
  float s2 = 0.0f;
#pragma unroll
  for (int i = 0; i < 4; ++i) { float d = v[i] - mu; s2 += d * d; }
  red[t] = s2; __syncthreads();
  for (int off = 64; off > 0; off >>= 1) {
    if (t < off) red[t] += red[t + off];
    __syncthreads();
  }
  float rstd = rsqrtf(red[0] * (1.0f / H_) + 1e-5f);
#pragma unroll
  for (int i = 0; i < 4; ++i) {
    int c = t + 128 * i;
    float o = (v[i] - mu) * rstd * g[c] + bta[c];
    outf[(long)r * H_ + c] = o;
    if (outb) outb[(long)r * H_ + c] = (bf16)o;
  }
}

// ---------------------------------------------------------------------------
// Host orchestration
// ---------------------------------------------------------------------------
extern "C" void kernel_launch(void* const* d_in, const int* in_sizes, int n_in,
                              void* d_out, int out_size, void* d_ws, size_t ws_size,
                              hipStream_t stream) {
  const float* x    = (const float*)d_in[0];
  const float* ea   = (const float*)d_in[1];
  const float* Wq   = (const float*)d_in[3];  const float* bq  = (const float*)d_in[4];
  const float* Wk   = (const float*)d_in[5];  const float* bk  = (const float*)d_in[6];
  const float* Wv   = (const float*)d_in[7];  const float* bv  = (const float*)d_in[8];
  const float* Wo   = (const float*)d_in[9];  const float* bo  = (const float*)d_in[10];
  const float* We1  = (const float*)d_in[11]; const float* be1 = (const float*)d_in[12];
  const float* We2  = (const float*)d_in[13]; const float* be2 = (const float*)d_in[14];
  const float* We3  = (const float*)d_in[15]; const float* be3 = (const float*)d_in[16];
  const float* Wp1  = (const float*)d_in[17]; const float* bp1 = (const float*)d_in[18];
  const float* Wp2  = (const float*)d_in[19]; const float* bp2 = (const float*)d_in[20];
  const float* sa_g = (const float*)d_in[21]; const float* sa_b = (const float*)d_in[22];
  const float* on_g = (const float*)d_in[23]; const float* on_b = (const float*)d_in[24];
  const float* alp  = (const float*)d_in[25]; const float* bet  = (const float*)d_in[26];
  float* out = (float*)d_out;

  char* p = (char*)d_ws;
  size_t off = 0;
  auto carve = [&](size_t bytes) { char* r = p + off; off += (bytes + 255) & ~(size_t)255; return r; };
  bf16*  wq_t   = (bf16*)carve((size_t)H_ * H_ * 2);
  bf16*  wk_t   = (bf16*)carve((size_t)H_ * H_ * 2);
  bf16*  wv_t   = (bf16*)carve((size_t)H_ * H_ * 2);
  bf16*  wo_t   = (bf16*)carve((size_t)H_ * H_ * 2);
  bf16*  we1_t  = (bf16*)carve((size_t)H_ * H_ * 2);
  bf16*  we2_t  = (bf16*)carve((size_t)H_ * H_ * 2);
  bf16*  wp1_t  = (bf16*)carve((size_t)H_ * 2 * H_ * 2);
  bf16*  wp2_t  = (bf16*)carve((size_t)2 * H_ * H_ * 2);
  bf16*  x_bf   = (bf16*)carve((size_t)MX_ * H_ * 2);
  float* coeffs = (float*)carve((size_t)MTOT_ * NH_ * 4);
  float* tmpf   = (float*)carve((size_t)MX_ * H_ * 4);
  bf16*  q_bf   = (bf16*)carve((size_t)MX_ * H_ * 2);
  bf16*  k_bf   = (bf16*)carve((size_t)MX_ * H_ * 2);
  bf16*  vT_bf  = (bf16*)carve((size_t)MX_ * H_ * 2);
  bf16*  at_bf  = (bf16*)carve((size_t)MX_ * H_ * 2);
  float* r1_f   = (float*)carve((size_t)MX_ * H_ * 4);
  bf16*  r1_bf  = (bf16*)carve((size_t)MX_ * H_ * 2);
  bf16*  ffh_bf = (bf16*)carve((size_t)MX_ * 2 * H_ * 2);

  const size_t edge_lds = 2 * 128 * LDA1 * sizeof(bf16);                       // 266,240 B
  const size_t attn_lds = 128 * SC_LD * sizeof(float) + 128 * PB_LD * sizeof(bf16); // 102,400 B
  (void)hipFuncSetAttribute((const void*)edge_mlp_kernel,
                            hipFuncAttributeMaxDynamicSharedMemorySize, (int)edge_lds);
  (void)hipFuncSetAttribute((const void*)attention_kernel,
                            hipFuncAttributeMaxDynamicSharedMemorySize, (int)attn_lds);

  // 1. weight transpose+convert and x convert
  const int HH = H_ * H_;
  wconv_kernel<<<(HH + 255) / 256, 256, 0, stream>>>(Wq, wq_t, H_, H_);
  wconv_kernel<<<(HH + 255) / 256, 256, 0, stream>>>(Wk, wk_t, H_, H_);
  wconv_kernel<<<(HH + 255) / 256, 256, 0, stream>>>(Wv, wv_t, H_, H_);
  wconv_kernel<<<(HH + 255) / 256, 256, 0, stream>>>(Wo, wo_t, H_, H_);
  wconv_kernel<<<(HH + 255) / 256, 256, 0, stream>>>(We1, we1_t, H_, H_);
  wconv_kernel<<<(HH + 255) / 256, 256, 0, stream>>>(We2, we2_t, H_, H_);
  wconv_kernel<<<(2 * HH + 255) / 256, 256, 0, stream>>>(Wp1, wp1_t, H_, 2 * H_);
  wconv_kernel<<<(2 * HH + 255) / 256, 256, 0, stream>>>(Wp2, wp2_t, 2 * H_, H_);
  cast_kernel<<<(MX_ * H_ + 255) / 256, 256, 0, stream>>>(x, x_bf, MX_ * H_);

  // 2. fused edge MLP (dominant work): 1024 workgroups, 260 KB LDS each
  edge_mlp_kernel<<<MTOT_ / 128, 256, edge_lds, stream>>>(ea, we1_t, be1, we2_t, be2,
                                                          We3, be3, coeffs);

  // 3. QKV projections + head reshapes
  dim3 g44(H_ / 128, MX_ / 128);
  gemm_bf16_kernel<<<g44, 256, 0, stream>>>(x_bf, wq_t, bq, tmpf, nullptr, MX_, H_, H_, 0);
  qkv_reshape_kernel<<<(MX_ * H_ + 255) / 256, 256, 0, stream>>>(tmpf, q_bf, 0.125f, 0);
  gemm_bf16_kernel<<<g44, 256, 0, stream>>>(x_bf, wk_t, bk, tmpf, nullptr, MX_, H_, H_, 0);
  qkv_reshape_kernel<<<(MX_ * H_ + 255) / 256, 256, 0, stream>>>(tmpf, k_bf, 1.0f, 0);
  gemm_bf16_kernel<<<g44, 256, 0, stream>>>(x_bf, wv_t, bv, tmpf, nullptr, MX_, H_, H_, 0);
  qkv_reshape_kernel<<<(MX_ * H_ + 255) / 256, 256, 0, stream>>>(tmpf, vT_bf, 1.0f, 1);

  // 4. fused attention (scores + bias + softmax + AV), one WG per (b,h)
  attention_kernel<<<BS_ * NH_, 256, attn_lds, stream>>>(q_bf, k_bf, vT_bf, coeffs,
                                                         alp, bet, at_bf);

  // 5. output projection + residual LN
  gemm_bf16_kernel<<<g44, 256, 0, stream>>>(at_bf, wo_t, bo, tmpf, nullptr, MX_, H_, H_, 0);
  ln_kernel<<<MX_, 128, 0, stream>>>(tmpf, x, sa_g, sa_b, r1_f, r1_bf);

  // 6. FFN (exact GELU) + final LN
  dim3 g88(2 * H_ / 128, MX_ / 128);
  gemm_bf16_kernel<<<g88, 256, 0, stream>>>(r1_bf, wp1_t, bp1, nullptr, ffh_bf,
                                            MX_, 2 * H_, H_, 2);
  gemm_bf16_kernel<<<g44, 256, 0, stream>>>(ffh_bf, wp2_t, bp2, tmpf, nullptr,
                                            MX_, H_, 2 * H_, 0);
  ln_kernel<<<MX_, 128, 0, stream>>>(tmpf, r1_f, on_g, on_b, out, nullptr);
}
